// PointNet2Backbone_30631706755379
// MI455X (gfx1250) — matrix operand order repaired and checked by reading, not compiled, not run on hardware
//
#include <hip/hip_runtime.h>

// ---------------------------------------------------------------------------
// PointNet++ backbone for MI455X (gfx1250, wave32, WMMA).
// All GEMMs use v_wmma_f32_16x16x32_bf16 (bf16 A/B, f32 accumulate).
// ---------------------------------------------------------------------------

typedef __attribute__((ext_vector_type(16))) __bf16 bf16x16;
typedef __attribute__((ext_vector_type(8)))  float  f32x8;

#define BB  4
#define NPT 4096
#define MM1 2048
#define MM2 512
#define MM3 51
#define KR  64

union BfVec { bf16x16 v; uint4 q[2]; };

__device__ __forceinline__ unsigned short f2bf_u(float f) {
  unsigned u = __float_as_uint(f);
  u += 0x7FFFu + ((u >> 16) & 1u);          // round-to-nearest-even
  return (unsigned short)(u >> 16);
}
__device__ __forceinline__ float bf2f(unsigned short h) {
  return __uint_as_float(((unsigned)h) << 16);
}

// ---------------------------------------------------------------------------
// Weight prep: W (K x N, f32, row-major) -> Wt (N x Kpad, bf16), zero-padded K.
// ---------------------------------------------------------------------------
__global__ void prep_weight_kernel(const float* __restrict__ W,
                                   unsigned short* __restrict__ Wt,
                                   int K, int N, int Kpad) {
  int i = blockIdx.x * blockDim.x + threadIdx.x;
  if (i >= N * Kpad) return;
  int n = i / Kpad, k = i - n * Kpad;
  float v = (k < K) ? W[k * N + n] : 0.0f;
  Wt[i] = f2bf_u(v);
}

// ---------------------------------------------------------------------------
// Farthest point sampling. One block per batch; dists live in LDS.
// Mirrors the jax scan: record current `far`, then update dists, then argmax.
// ---------------------------------------------------------------------------
__global__ __launch_bounds__(256) void fps_kernel(const float* __restrict__ pos,
                                                  int n, int m, int* __restrict__ samp) {
  __shared__ float dists[NPT];
  __shared__ float rv[256];
  __shared__ int   ri[256];
  __shared__ int   s_far;
  const float* p = pos + (long)blockIdx.x * n * 3;
  int* out = samp + (long)blockIdx.x * m;
  int tid = threadIdx.x;
  for (int i = tid; i < n; i += 256) dists[i] = 1e10f;
  if (tid == 0) s_far = 0;
  __syncthreads();
  for (int s = 0; s < m; s++) {
    int far = s_far;
    if (tid == 0) out[s] = far;
    float fx = p[far * 3 + 0], fy = p[far * 3 + 1], fz = p[far * 3 + 2];
    float best = -1.0f; int bi = 0x7fffffff;
    for (int i = tid; i < n; i += 256) {
      float dx = p[i * 3 + 0] - fx, dy = p[i * 3 + 1] - fy, dz = p[i * 3 + 2] - fz;
      float dd = fminf(dists[i], dx * dx + dy * dy + dz * dz);
      dists[i] = dd;
      if (dd > best) { best = dd; bi = i; }
    }
    rv[tid] = best; ri[tid] = bi;
    __syncthreads();
    for (int off = 128; off > 0; off >>= 1) {
      if (tid < off) {
        if (rv[tid + off] > rv[tid] ||
            (rv[tid + off] == rv[tid] && ri[tid + off] < ri[tid])) {
          rv[tid] = rv[tid + off]; ri[tid] = ri[tid + off];
        }
      }
      __syncthreads();
    }
    if (tid == 0) s_far = ri[0];
    __syncthreads();
  }
}

__global__ void gather_pos_kernel(const float* __restrict__ pos_src,
                                  const int* __restrict__ samp,
                                  float* __restrict__ pos_dst,
                                  int n_src, int m, int total) {
  int i = blockIdx.x * blockDim.x + threadIdx.x;
  if (i >= total) return;
  int b = i / m, j = i - b * m;
  int s = samp[b * m + j];
  const float* p = pos_src + (long)(b * n_src + s) * 3;
  float* o = pos_dst + (long)i * 3;
  o[0] = p[0]; o[1] = p[1]; o[2] = p[2];
}

// ---------------------------------------------------------------------------
// Radius grouping: first KR in-radius indices (index order), padded with the
// lowest out-of-radius indices (matches the stable-argsort reference).
// ---------------------------------------------------------------------------
__global__ void radius_kernel(const float* __restrict__ pos_src,
                              const float* __restrict__ pos_dst,
                              int n_src, int m, float r2,
                              int* __restrict__ idx, int* __restrict__ msk,
                              int total) {
  int d = blockIdx.x * blockDim.x + threadIdx.x;
  if (d >= total) return;
  int b = d / m;
  const float* ps = pos_src + (long)b * n_src * 3;
  float px = pos_dst[d * 3 + 0], py = pos_dst[d * 3 + 1], pz = pos_dst[d * 3 + 2];
  int* oi = idx + (long)d * KR;
  int* om = msk + (long)d * KR;
  int cnt = 0;
  for (int s = 0; s < n_src && cnt < KR; s++) {
    float dx = ps[s * 3 + 0] - px, dy = ps[s * 3 + 1] - py, dz = ps[s * 3 + 2] - pz;
    float dd = dx * dx + dy * dy + dz * dz;
    if (dd < r2) { oi[cnt] = s; om[cnt] = 1; cnt++; }
  }
  if (cnt < KR) {
    for (int s = 0; s < n_src && cnt < KR; s++) {
      float dx = ps[s * 3 + 0] - px, dy = ps[s * 3 + 1] - py, dz = ps[s * 3 + 2] - pz;
      float dd = dx * dx + dy * dy + dz * dz;
      if (!(dd < r2)) { oi[cnt] = s; om[cnt] = 0; cnt++; }
    }
  }
}

// ---------------------------------------------------------------------------
// One MLP layer, LDS->LDS, via v_wmma_f32_16x16x32_bf16.
// A: 64 x K bf16 in LDS (row-major, K multiple of 32).
// Wt: N x K bf16 in global (row-major, B fragments in the same k order as A).
// 4 waves sweep the 4 x (N/16) output tiles.
// Fragment layouts per CDNA5 ISA 7.12.2.
// ---------------------------------------------------------------------------
__device__ __forceinline__ void sa_layer(const unsigned short* Asrc,
                                         unsigned short* Odst,
                                         const unsigned short* __restrict__ Wt,
                                         const float* __restrict__ bias,
                                         int K, int N, bool relu) {
  int tid  = threadIdx.x;
  int wave = tid >> 5, lane = tid & 31;
  int lrow = lane & 15, hi = lane >> 4;
  int khalf = hi ? 8 : 0;    // A: lanes 16-31 hold K {8..15, 24..31}
  int kbb   = hi ? 16 : 0;   // unused; B also uses khalf: lanes 16-31 hold K {8..15, 24..31}
  int nTiles = 4 * (N >> 4);
  for (int t = wave; t < nTiles; t += 4) {
    int rowTile = t & 3, colTile = t >> 2;
    int col = colTile * 16 + lrow;
    const unsigned short* Ar = Asrc + (rowTile * 16 + lrow) * K;
    const unsigned short* Br = Wt + (long)col * K;
    f32x8 acc = {};
    for (int kc = 0; kc < K; kc += 32) {
      BfVec av, wv;
      av.q[0] = *(const uint4*)(Ar + kc + khalf);
      av.q[1] = *(const uint4*)(Ar + kc + 16 + khalf);
      wv.q[0] = *(const uint4*)(Br + kc + khalf);
      wv.q[1] = *(const uint4*)(Br + kc + 16 + khalf);
      acc = __builtin_amdgcn_wmma_f32_16x16x32_bf16(
          false, av.v, false, wv.v, (short)0, acc, false, false);
    }
    float bz = bias[col];
    int rbase = rowTile * 16 + (hi ? 8 : 0);
#pragma unroll
    for (int i = 0; i < 8; i++) {
      float v = acc[i] + bz;
      if (relu) v = fmaxf(v, 0.0f);
      Odst[(rbase + i) * N + col] = f2bf_u(v);
    }
  }
}

// ---------------------------------------------------------------------------
// Fused SA: one workgroup per dst point. Gather 64 neighbors' [x_j, rel_pos]
// into LDS (bf16), run 3 WMMA MLP layers LDS<->LDS, masked max-pool to x_out.
// ---------------------------------------------------------------------------
__global__ __launch_bounds__(128) void sa_kernel(
    const float* __restrict__ x_src, const float* __restrict__ pos_src,
    const float* __restrict__ pos_dst,
    const int* __restrict__ nbr_idx, const int* __restrict__ nbr_msk,
    const unsigned short* __restrict__ Wt0, const float* __restrict__ bs0,
    const unsigned short* __restrict__ Wt1, const float* __restrict__ bs1,
    const unsigned short* __restrict__ Wt2, const float* __restrict__ bs2,
    int C_in, int K0pad, int N0, int N1, int N2,
    int n_src, int m, int maxw, float* __restrict__ x_out) {
  extern __shared__ unsigned short smem[];
  unsigned short* bufA = smem;
  unsigned short* bufB = smem + 64 * maxw;
  __shared__ int s_idx[KR];
  __shared__ int s_msk[KR];
  int tid = threadIdx.x;
  int d = blockIdx.x;
  int b = d / m;
  float pdx = pos_dst[d * 3 + 0], pdy = pos_dst[d * 3 + 1], pdz = pos_dst[d * 3 + 2];
  if (tid < KR) {
    s_idx[tid] = nbr_idx[(long)d * KR + tid];
    s_msk[tid] = nbr_msk[(long)d * KR + tid];
  }
  __syncthreads();
  int Cp3 = C_in + 3;
  for (int e = tid; e < KR * K0pad; e += 128) {
    int j = e / K0pad, c = e - j * K0pad;
    int src = b * n_src + s_idx[j];
    float v = 0.0f;
    if (c < C_in) {
      v = x_src[(long)src * C_in + c];
    } else if (c < Cp3) {
      int a = c - C_in;
      v = pos_src[(long)src * 3 + a] - (a == 0 ? pdx : (a == 1 ? pdy : pdz));
    }
    bufA[e] = f2bf_u(v);
  }
  __syncthreads();
  sa_layer(bufA, bufB, Wt0, bs0, K0pad, N0, true);
  __syncthreads();
  sa_layer(bufB, bufA, Wt1, bs1, N0, N1, true);
  __syncthreads();
  sa_layer(bufA, bufB, Wt2, bs2, N1, N2, false);
  __syncthreads();
  for (int c = tid; c < N2; c += 128) {
    float mx = -3.0e38f; int any = 0;
    for (int j = 0; j < KR; j++) {
      if (s_msk[j]) { any = 1; mx = fmaxf(mx, bf2f(bufB[j * N2 + c])); }
    }
    x_out[(long)d * N2 + c] = any ? mx : 0.0f;
  }
}

// ---------------------------------------------------------------------------
// 3-NN (inverse-distance weights) for feature propagation.
// ---------------------------------------------------------------------------
__global__ void knn3_kernel(const float* __restrict__ pos_src,
                            const float* __restrict__ pos_dst,
                            int n_src, int m_dst, int total,
                            int* __restrict__ kidx, float* __restrict__ kw) {
  int d = blockIdx.x * blockDim.x + threadIdx.x;
  if (d >= total) return;
  int b = d / m_dst;
  const float* ps = pos_src + (long)b * n_src * 3;
  float px = pos_dst[d * 3 + 0], py = pos_dst[d * 3 + 1], pz = pos_dst[d * 3 + 2];
  float d0 = 3.0e38f, d1 = 3.0e38f, d2 = 3.0e38f;
  int i0 = 0, i1 = 0, i2 = 0;
  for (int s = 0; s < n_src; s++) {
    float dx = ps[s * 3 + 0] - px, dy = ps[s * 3 + 1] - py, dz = ps[s * 3 + 2] - pz;
    float dd = dx * dx + dy * dy + dz * dz;
    if (dd < d0)      { d2 = d1; i2 = i1; d1 = d0; i1 = i0; d0 = dd; i0 = s; }
    else if (dd < d1) { d2 = d1; i2 = i1; d1 = dd; i1 = s; }
    else if (dd < d2) { d2 = dd; i2 = s; }
  }
  kidx[d * 3 + 0] = i0; kidx[d * 3 + 1] = i1; kidx[d * 3 + 2] = i2;
  kw[d * 3 + 0] = 1.0f / (sqrtf(fmaxf(d0, 1e-12f)) + 1e-8f);
  kw[d * 3 + 1] = 1.0f / (sqrtf(fmaxf(d1, 1e-12f)) + 1e-8f);
  kw[d * 3 + 2] = 1.0f / (sqrtf(fmaxf(d2, 1e-12f)) + 1e-8f);
}

// ---------------------------------------------------------------------------
// Build bf16 FP feature matrix: [knn-interp(h_src), skip] padded to Kpad.
// ---------------------------------------------------------------------------
__global__ void fp_feat_kernel(const float* __restrict__ h_src, int C_interp,
                               const float* __restrict__ skip, int C_skip,
                               const int* __restrict__ kidx, const float* __restrict__ kw,
                               int n_src, int m_dst,
                               unsigned short* __restrict__ F, int Kpad, int total) {
  int e = blockIdx.x * blockDim.x + threadIdx.x;
  if (e >= total) return;
  int r = e / Kpad, c = e - r * Kpad;
  float v = 0.0f;
  if (c < C_interp) {
    int b = r / m_dst;
    float num = 0.0f, den = 0.0f;
#pragma unroll
    for (int k = 0; k < 3; k++) {
      float w = kw[r * 3 + k];
      int s = kidx[r * 3 + k];
      num += w * h_src[(long)(b * n_src + s) * C_interp + c];
      den += w;
    }
    v = num / (den + 1e-8f);
  } else if (c < C_interp + C_skip) {
    v = skip[(long)r * C_skip + (c - C_interp)];
  }
  F[e] = f2bf_u(v);
}

// ---------------------------------------------------------------------------
// Dense WMMA GEMM: out = act(A @ Wt^T + bias). A: rows x K bf16, Wt: N x K
// bf16. One wave per 16x16 tile, 8 waves per block.
// ---------------------------------------------------------------------------
__global__ __launch_bounds__(256) void gemm_bf16_kernel(
    const unsigned short* __restrict__ A, const unsigned short* __restrict__ Wt,
    const float* __restrict__ bias, void* __restrict__ out,
    int rows, int K, int N, int relu, int out_f32, int ldo) {
  int wave = threadIdx.x >> 5, lane = threadIdx.x & 31;
  int colTiles = N >> 4;
  int totTiles = (rows >> 4) * colTiles;
  int tile = blockIdx.x * 8 + wave;
  if (tile >= totTiles) return;
  int rowTile = tile / colTiles;
  int colTile = tile - rowTile * colTiles;
  int lrow = lane & 15, hi = lane >> 4;
  int khalf = hi ? 8 : 0;
  int kbb   = hi ? 16 : 0;
  int col = colTile * 16 + lrow;
  const unsigned short* Ar = A + (long)(rowTile * 16 + lrow) * K;
  const unsigned short* Br = Wt + (long)col * K;
  f32x8 acc = {};
  for (int kc = 0; kc < K; kc += 32) {
    BfVec av, wv;
    av.q[0] = *(const uint4*)(Ar + kc + khalf);
    av.q[1] = *(const uint4*)(Ar + kc + 16 + khalf);
    wv.q[0] = *(const uint4*)(Br + kc + khalf);
    wv.q[1] = *(const uint4*)(Br + kc + 16 + khalf);
    acc = __builtin_amdgcn_wmma_f32_16x16x32_bf16(
        false, av.v, false, wv.v, (short)0, acc, false, false);
  }
  float bz = bias[col];
  int rbase = rowTile * 16 + (hi ? 8 : 0);
#pragma unroll
  for (int i = 0; i < 8; i++) {
    float v = acc[i] + bz;
    if (relu) v = fmaxf(v, 0.0f);
    if (out_f32) ((float*)out)[(long)(rbase + i) * ldo + col] = v;
    else ((unsigned short*)out)[(long)(rbase + i) * ldo + col] = f2bf_u(v);
  }
}

// ---------------------------------------------------------------------------
// Global feature: column-wise max over the MM3 coarsest points.
// ---------------------------------------------------------------------------
__global__ void gfeat_kernel(const float* __restrict__ x3, float* __restrict__ out,
                             int m, int C, int total) {
  int i = blockIdx.x * blockDim.x + threadIdx.x;
  if (i >= total) return;
  int b = i / C, c = i - b * C;
  float mx = -3.0e38f;
  for (int j = 0; j < m; j++) mx = fmaxf(mx, x3[(long)(b * m + j) * C + c]);
  out[i] = mx;
}

// ---------------------------------------------------------------------------
// Host orchestration.
// ---------------------------------------------------------------------------
static inline int cdiv(long a, long b) { return (int)((a + b - 1) / b); }

extern "C" void kernel_launch(void* const* d_in, const int* in_sizes, int n_in,
                              void* d_out, int out_size, void* d_ws, size_t ws_size,
                              hipStream_t stream) {
  const float* x0   = (const float*)d_in[0];
  const float* pos0 = (const float*)d_in[1];
  const float* sa1W0 = (const float*)d_in[3];  const float* sa1b0 = (const float*)d_in[4];
  const float* sa1W1 = (const float*)d_in[5];  const float* sa1b1 = (const float*)d_in[6];
  const float* sa1W2 = (const float*)d_in[7];  const float* sa1b2 = (const float*)d_in[8];
  const float* sa2W0 = (const float*)d_in[9];  const float* sa2b0 = (const float*)d_in[10];
  const float* sa2W1 = (const float*)d_in[11]; const float* sa2b1 = (const float*)d_in[12];
  const float* sa2W2 = (const float*)d_in[13]; const float* sa2b2 = (const float*)d_in[14];
  const float* sa3W0 = (const float*)d_in[15]; const float* sa3b0 = (const float*)d_in[16];
  const float* sa3W1 = (const float*)d_in[17]; const float* sa3b1 = (const float*)d_in[18];
  const float* sa3W2 = (const float*)d_in[19]; const float* sa3b2 = (const float*)d_in[20];
  const float* fp3W0 = (const float*)d_in[21]; const float* fp3b0 = (const float*)d_in[22];
  const float* fp3W1 = (const float*)d_in[23]; const float* fp3b1 = (const float*)d_in[24];
  const float* fp2W0 = (const float*)d_in[25]; const float* fp2b0 = (const float*)d_in[26];
  const float* fp2W1 = (const float*)d_in[27]; const float* fp2b1 = (const float*)d_in[28];
  const float* fp1W0 = (const float*)d_in[29]; const float* fp1b0 = (const float*)d_in[30];
  const float* fp1W1 = (const float*)d_in[31]; const float* fp1b1 = (const float*)d_in[32];

  char* wsb = (char*)d_ws;
  size_t off = 0;
  auto alloc = [&](size_t bytes) -> void* {
    void* p = (void*)(wsb + off);
    off = (off + bytes + 255) & ~((size_t)255);
    return p;
  };

  // bf16 transposed/padded weights
  unsigned short* t_sa1_0 = (unsigned short*)alloc((size_t)64  * 160 * 2);
  unsigned short* t_sa1_1 = (unsigned short*)alloc((size_t)64  * 64  * 2);
  unsigned short* t_sa1_2 = (unsigned short*)alloc((size_t)128 * 64  * 2);
  unsigned short* t_sa2_0 = (unsigned short*)alloc((size_t)128 * 160 * 2);
  unsigned short* t_sa2_1 = (unsigned short*)alloc((size_t)128 * 128 * 2);
  unsigned short* t_sa2_2 = (unsigned short*)alloc((size_t)256 * 128 * 2);
  unsigned short* t_sa3_0 = (unsigned short*)alloc((size_t)256 * 288 * 2);
  unsigned short* t_sa3_1 = (unsigned short*)alloc((size_t)512 * 256 * 2);
  unsigned short* t_sa3_2 = (unsigned short*)alloc((size_t)512 * 512 * 2);
  unsigned short* t_fp3_0 = (unsigned short*)alloc((size_t)256 * 768 * 2);
  unsigned short* t_fp3_1 = (unsigned short*)alloc((size_t)256 * 256 * 2);
  unsigned short* t_fp2_0 = (unsigned short*)alloc((size_t)256 * 384 * 2);
  unsigned short* t_fp2_1 = (unsigned short*)alloc((size_t)128 * 256 * 2);
  unsigned short* t_fp1_0 = (unsigned short*)alloc((size_t)128 * 256 * 2);
  unsigned short* t_fp1_1 = (unsigned short*)alloc((size_t)128 * 128 * 2);

  int*   samp1 = (int*)alloc((size_t)BB * MM1 * 4);
  int*   samp2 = (int*)alloc((size_t)BB * MM2 * 4);
  int*   samp3 = (int*)alloc((size_t)BB * MM3 * 4);
  float* pos1  = (float*)alloc((size_t)BB * MM1 * 3 * 4);
  float* pos2  = (float*)alloc((size_t)BB * MM2 * 3 * 4);
  float* pos3  = (float*)alloc((size_t)BB * MM3 * 3 * 4);
  int*   idx1  = (int*)alloc((size_t)BB * MM1 * KR * 4);
  int*   msk1  = (int*)alloc((size_t)BB * MM1 * KR * 4);
  int*   idx2  = (int*)alloc((size_t)BB * MM2 * KR * 4);
  int*   msk2  = (int*)alloc((size_t)BB * MM2 * KR * 4);
  int*   idx3  = (int*)alloc((size_t)BB * MM3 * KR * 4);
  int*   msk3  = (int*)alloc((size_t)BB * MM3 * KR * 4);
  float* x1    = (float*)alloc((size_t)BB * MM1 * 128 * 4);
  float* x2    = (float*)alloc((size_t)BB * MM2 * 256 * 4);
  float* x3    = (float*)alloc((size_t)BB * MM3 * 512 * 4);
  int*   ki3   = (int*)alloc((size_t)BB * MM2 * 3 * 4);
  float* kw3   = (float*)alloc((size_t)BB * MM2 * 3 * 4);
  int*   ki2   = (int*)alloc((size_t)BB * MM1 * 3 * 4);
  float* kw2   = (float*)alloc((size_t)BB * MM1 * 3 * 4);
  int*   ki1   = (int*)alloc((size_t)BB * NPT * 3 * 4);
  float* kw1   = (float*)alloc((size_t)BB * NPT * 3 * 4);
  unsigned short* F3 = (unsigned short*)alloc((size_t)BB * MM2 * 768 * 2);
  unsigned short* H3 = (unsigned short*)alloc((size_t)BB * MM2 * 256 * 2);
  float*          h3 = (float*)alloc((size_t)BB * MM2 * 256 * 4);
  unsigned short* F2 = (unsigned short*)alloc((size_t)BB * MM1 * 384 * 2);
  unsigned short* H2 = (unsigned short*)alloc((size_t)BB * MM1 * 256 * 2);
  float*          h2 = (float*)alloc((size_t)BB * MM1 * 128 * 4);
  unsigned short* F1 = (unsigned short*)alloc((size_t)BB * NPT * 256 * 2);
  unsigned short* H1 = (unsigned short*)alloc((size_t)BB * NPT * 128 * 2);

  auto prep = [&](const float* W, unsigned short* Wt, int K, int N, int Kpad) {
    int tot = N * Kpad;
    prep_weight_kernel<<<cdiv(tot, 256), 256, 0, stream>>>(W, Wt, K, N, Kpad);
  };
  prep(sa1W0, t_sa1_0, 131, 64, 160);
  prep(sa1W1, t_sa1_1, 64, 64, 64);
  prep(sa1W2, t_sa1_2, 64, 128, 64);
  prep(sa2W0, t_sa2_0, 131, 128, 160);
  prep(sa2W1, t_sa2_1, 128, 128, 128);
  prep(sa2W2, t_sa2_2, 128, 256, 128);
  prep(sa3W0, t_sa3_0, 259, 256, 288);
  prep(sa3W1, t_sa3_1, 256, 512, 256);
  prep(sa3W2, t_sa3_2, 512, 512, 512);
  prep(fp3W0, t_fp3_0, 768, 256, 768);
  prep(fp3W1, t_fp3_1, 256, 256, 256);
  prep(fp2W0, t_fp2_0, 384, 256, 384);
  prep(fp2W1, t_fp2_1, 256, 128, 256);
  prep(fp1W0, t_fp1_0, 256, 128, 256);
  prep(fp1W1, t_fp1_1, 128, 128, 128);

  // ---------------- SA level 1 ----------------
  fps_kernel<<<BB, 256, 0, stream>>>(pos0, NPT, MM1, samp1);
  gather_pos_kernel<<<cdiv(BB * MM1, 256), 256, 0, stream>>>(pos0, samp1, pos1, NPT, MM1, BB * MM1);
  radius_kernel<<<cdiv(BB * MM1, 256), 256, 0, stream>>>(pos0, pos1, NPT, MM1, 100.0f, idx1, msk1, BB * MM1);
  sa_kernel<<<BB * MM1, 128, (size_t)2 * 64 * 160 * 2, stream>>>(
      x0, pos0, pos1, idx1, msk1,
      t_sa1_0, sa1b0, t_sa1_1, sa1b1, t_sa1_2, sa1b2,
      128, 160, 64, 64, 128, NPT, MM1, 160, x1);

  // ---------------- SA level 2 ----------------
  fps_kernel<<<BB, 256, 0, stream>>>(pos1, MM1, MM2, samp2);
  gather_pos_kernel<<<cdiv(BB * MM2, 256), 256, 0, stream>>>(pos1, samp2, pos2, MM1, MM2, BB * MM2);
  radius_kernel<<<cdiv(BB * MM2, 256), 256, 0, stream>>>(pos1, pos2, MM1, MM2, 400.0f, idx2, msk2, BB * MM2);
  sa_kernel<<<BB * MM2, 128, (size_t)2 * 64 * 256 * 2, stream>>>(
      x1, pos1, pos2, idx2, msk2,
      t_sa2_0, sa2b0, t_sa2_1, sa2b1, t_sa2_2, sa2b2,
      128, 160, 128, 128, 256, MM1, MM2, 256, x2);

  // ---------------- SA level 3 ----------------
  fps_kernel<<<BB, 256, 0, stream>>>(pos2, MM2, MM3, samp3);
  gather_pos_kernel<<<cdiv(BB * MM3, 256), 256, 0, stream>>>(pos2, samp3, pos3, MM2, MM3, BB * MM3);
  radius_kernel<<<cdiv(BB * MM3, 256), 256, 0, stream>>>(pos2, pos3, MM2, MM3, 1600.0f, idx3, msk3, BB * MM3);
  sa_kernel<<<BB * MM3, 128, (size_t)2 * 64 * 512 * 2, stream>>>(
      x2, pos2, pos3, idx3, msk3,
      t_sa3_0, sa3b0, t_sa3_1, sa3b1, t_sa3_2, sa3b2,
      256, 288, 256, 512, 512, MM2, MM3, 512, x3);

  // ---------------- global feature -> tail of d_out ----------------
  float* gfeat = (float*)d_out + (size_t)BB * NPT * 128;
  gfeat_kernel<<<cdiv(BB * 512, 256), 256, 0, stream>>>(x3, gfeat, MM3, 512, BB * 512);

  // ---------------- FP level 3 (p3 -> p2) ----------------
  knn3_kernel<<<cdiv(BB * MM2, 256), 256, 0, stream>>>(pos3, pos2, MM3, MM2, BB * MM2, ki3, kw3);
  fp_feat_kernel<<<cdiv((long)BB * MM2 * 768, 256), 256, 0, stream>>>(
      x3, 512, x2, 256, ki3, kw3, MM3, MM2, F3, 768, BB * MM2 * 768);
  gemm_bf16_kernel<<<cdiv((BB * MM2 / 16) * (256 / 16), 8), 256, 0, stream>>>(
      F3, t_fp3_0, fp3b0, H3, BB * MM2, 768, 256, 1, 0, 256);
  gemm_bf16_kernel<<<cdiv((BB * MM2 / 16) * (256 / 16), 8), 256, 0, stream>>>(
      H3, t_fp3_1, fp3b1, h3, BB * MM2, 256, 256, 0, 1, 256);

  // ---------------- FP level 2 (p2 -> p1) ----------------
  knn3_kernel<<<cdiv(BB * MM1, 256), 256, 0, stream>>>(pos2, pos1, MM2, MM1, BB * MM1, ki2, kw2);
  fp_feat_kernel<<<cdiv((long)BB * MM1 * 384, 256), 256, 0, stream>>>(
      h3, 256, x1, 128, ki2, kw2, MM2, MM1, F2, 384, BB * MM1 * 384);
  gemm_bf16_kernel<<<cdiv((BB * MM1 / 16) * (256 / 16), 8), 256, 0, stream>>>(
      F2, t_fp2_0, fp2b0, H2, BB * MM1, 384, 256, 1, 0, 256);
  gemm_bf16_kernel<<<cdiv((BB * MM1 / 16) * (128 / 16), 8), 256, 0, stream>>>(
      H2, t_fp2_1, fp2b1, h2, BB * MM1, 256, 128, 0, 1, 128);

  // ---------------- FP level 1 (p1 -> p0) ----------------
  knn3_kernel<<<cdiv(BB * NPT, 256), 256, 0, stream>>>(pos1, pos0, MM1, NPT, BB * NPT, ki1, kw1);
  fp_feat_kernel<<<cdiv((long)BB * NPT * 256, 256), 256, 0, stream>>>(
      h2, 128, x0, 128, ki1, kw1, MM1, NPT, F1, 256, BB * NPT * 256);
  gemm_bf16_kernel<<<cdiv((BB * NPT / 16) * (128 / 16), 8), 256, 0, stream>>>(
      F1, t_fp1_0, fp1b0, H1, BB * NPT, 256, 128, 1, 0, 128);
  gemm_bf16_kernel<<<cdiv((BB * NPT / 16) * (128 / 16), 8), 256, 0, stream>>>(
      H1, t_fp1_1, fp1b1, d_out, BB * NPT, 128, 128, 0, 1, 128);
}
